// RouteNet_Fermi_9139690406020
// MI455X (gfx1250) — matrix operand order repaired and checked by reading, not compiled, not run on hardware
//
#include <hip/hip_runtime.h>

#define P_N 80000
#define T_N 8
#define L_N 10000
#define Q_N 20000
#define MPL 80
#define MPQ 40
#define MQL 3
#define D_N 32
#define ITERS 8

typedef __attribute__((ext_vector_type(16))) __bf16 v16bf;
typedef __attribute__((ext_vector_type(8)))  float  v8f;

// bit-exact RNE f32->bf16 (used only in the one-shot weight pack kernel)
__device__ __forceinline__ unsigned short f2bf_bits(float f) {
  unsigned int x = __float_as_uint(f);
  x += 0x7FFFu + ((x >> 16) & 1u);
  return (unsigned short)(x >> 16);
}

__device__ __forceinline__ v8f wmma_bf16(v16bf a, v16bf b, v8f c) {
  // D = A(16x32 bf16) x B(32x16 bf16) + C(16x16 f32)
  return __builtin_amdgcn_wmma_f32_16x16x32_bf16(false, a, false, b,
                                                 (short)0, c, false, false);
}

// fast transcendentals: v_exp_f32 + v_rcp_f32 (TRANS ops, co-execute with WMMA)
__device__ __forceinline__ float sigmf(float x) {
  return __builtin_amdgcn_rcpf(1.0f + __expf(-x));
}
__device__ __forceinline__ float tanh_fast(float x) {
  return 1.0f - 2.0f * __builtin_amdgcn_rcpf(__expf(2.0f * x) + 1.0f);
}

// Build this lane's share of a 16x32 bf16 A-fragment from one 32-float row.
// A layout (ISA 7.12.2): lane m = l%16, khalf = l/16 ; VGPR0-3 hold K = 8*khalf+0..7,
// VGPR4-7 hold K = 16+8*khalf+0..7 (2 bf16 per VGPR). Native cvt -> v_cvt_pk_bf16_f32.
__device__ __forceinline__ v16bf gatherA(const float* __restrict__ row, int khalf) {
  const float4* r1 = (const float4*)(row + 8 * khalf);
  const float4* r2 = (const float4*)(row + 16 + 8 * khalf);
  float4 a0 = r1[0], a1 = r1[1], b0 = r2[0], b1 = r2[1];
  float va[16] = {a0.x,a0.y,a0.z,a0.w,a1.x,a1.y,a1.z,a1.w,
                  b0.x,b0.y,b0.z,b0.w,b1.x,b1.y,b1.z,b1.w};
  v16bf a;
#pragma unroll
  for (int i = 0; i < 16; i++) a[i] = (__bf16)va[i];
  return a;
}

// GRU gates in C/D layout. Tiles 0..5 cover gate columns [16j,16j+16) of the 96-wide
// pre-activation: z=tiles{0,1}, r=tiles{2,3}, h=tiles{4,5}. h0/h1 = units nlo / 16+nlo.
__device__ __forceinline__ void gru_gates(const v8f* xg, const v8f* hg,
                                          const float* bi6, const float* bh6,
                                          int r, float h0, float h1,
                                          float& o0, float& o1) {
  float z0 = sigmf(xg[0][r] + bi6[0] + hg[0][r] + bh6[0]);
  float r0 = sigmf(xg[2][r] + bi6[2] + hg[2][r] + bh6[2]);
  float c0 = tanh_fast(xg[4][r] + bi6[4] + r0 * (hg[4][r] + bh6[4]));
  o0 = z0 * h0 + (1.0f - z0) * c0;
  float z1 = sigmf(xg[1][r] + bi6[1] + hg[1][r] + bh6[1]);
  float r1 = sigmf(xg[3][r] + bi6[3] + hg[3][r] + bh6[3]);
  float c1 = tanh_fast(xg[5][r] + bi6[5] + r1 * (hg[5][r] + bh6[5]));
  o1 = z1 * h1 + (1.0f - z1) * c1;
}

// ---------------------------------------------------------------------------
// Pack an [in_dim, 96] fp32 weight matrix into bf16 B-fragments.
// B layout (ISA 7.12.2): lane n = l%16, khalf = l/16; VGPR v holds
// K = 32c + 8*khalf + 2*(v&3) + 16*(v>>2) (lo 16b) and +1 (hi 16b).
// dst dword index = ((c*6 + j)*32 + lane)*8 + v -> each lane's frag = 32 contiguous B.
// ---------------------------------------------------------------------------
__global__ void __launch_bounds__(256) pack_b_kernel(const float* __restrict__ W,
                                                     unsigned int* __restrict__ dst,
                                                     int chunks) {
  int tid = blockIdx.x * 256 + threadIdx.x;
  int total = chunks * 6 * 32 * 8;
  if (tid >= total) return;
  int v = tid & 7;
  int lane = (tid >> 3) & 31;
  int j = (tid >> 8) % 6;
  int c = (tid >> 8) / 6;
  int n = 16 * j + (lane & 15);
  int k0 = 32 * c + 8 * (lane >> 4) + 2 * (v & 3) + 16 * (v >> 2);
  unsigned int lo = f2bf_bits(W[k0 * 96 + n]);
  unsigned int hi = f2bf_bits(W[(k0 + 1) * 96 + n]);
  dst[tid] = lo | (hi << 16);
}

// ---------------------------------------------------------------------------
// Embedding MLPs (tiny FLOPs -> plain VALU, weights staged in LDS)
// ---------------------------------------------------------------------------
__global__ void __launch_bounds__(256) path_emb_kernel(
    const float* __restrict__ traffic, const float* __restrict__ packets,
    const float* __restrict__ eq_lambda, const float* __restrict__ avg_pkts_lambda,
    const float* __restrict__ exp_max_factor, const float* __restrict__ pkts_lambda_on,
    const float* __restrict__ avg_t_off, const float* __restrict__ avg_t_on,
    const float* __restrict__ ar_a, const float* __restrict__ sigma,
    const int* __restrict__ model,
    const float* __restrict__ W1, const float* __restrict__ b1,
    const float* __restrict__ W2, const float* __restrict__ b2,
    float* __restrict__ path_state) {
  __shared__ float sW1[17 * 32], sW2[32 * 32], sb1[32], sb2[32];
  for (int i = threadIdx.x; i < 17 * 32; i += 256) sW1[i] = W1[i];
  for (int i = threadIdx.x; i < 32 * 32; i += 256) sW2[i] = W2[i];
  if (threadIdx.x < 32) { sb1[threadIdx.x] = b1[threadIdx.x]; sb2[threadIdx.x] = b2[threadIdx.x]; }
  __syncthreads();
  int p = blockIdx.x * 256 + threadIdx.x;
  if (p >= P_N) return;
  float f[17];
  f[0] = (traffic[p] - 1385.4058837890625f) / 859.8118896484375f;
  f[1] = (packets[p] - 1.4015231132507324f) / 0.8932565450668335f;
  int m = model[p];
#pragma unroll
  for (int i = 0; i < 7; i++) f[2 + i] = (i == m) ? 1.0f : 0.0f;
  f[9]  = (eq_lambda[p] - 1350.97119140625f) / 858.316162109375f;
  f[10] = (avg_pkts_lambda[p] - 0.9117304086685181f) / 0.9723503589630127f;
  f[11] = (exp_max_factor[p] - 6.663637638092041f) / 4.715115070343018f;
  f[12] = (pkts_lambda_on[p] - 0.9116322994232178f) / 1.651275396347046f;
  f[13] = (avg_t_off[p] - 1.6649284362792969f) / 2.356407403945923f;
  f[14] = (avg_t_on[p] - 1.6649284362792969f) / 2.356407403945923f;
  f[15] = ar_a[p];
  f[16] = sigma[p];
  float h1[32];
#pragma unroll
  for (int j = 0; j < 32; j++) {
    float s = sb1[j];
    for (int i = 0; i < 17; i++) s += f[i] * sW1[i * 32 + j];
    h1[j] = fmaxf(s, 0.0f);
  }
#pragma unroll
  for (int j = 0; j < 32; j++) {
    float s = sb2[j];
    for (int i = 0; i < 32; i++) s += h1[i] * sW2[i * 32 + j];
    path_state[p * 32 + j] = fmaxf(s, 0.0f);
  }
}

__global__ void __launch_bounds__(256) link_emb_kernel(
    const float* __restrict__ traffic, const float* __restrict__ capacity,
    const int* __restrict__ policy, const int* __restrict__ ptl,
    const float* __restrict__ W1, const float* __restrict__ b1,
    const float* __restrict__ W2, const float* __restrict__ b2,
    float* __restrict__ link_state) {
  __shared__ float sW1[5 * 32], sW2[32 * 32], sb1[32], sb2[32];
  for (int i = threadIdx.x; i < 5 * 32; i += 256) sW1[i] = W1[i];
  for (int i = threadIdx.x; i < 32 * 32; i += 256) sW2[i] = W2[i];
  if (threadIdx.x < 32) { sb1[threadIdx.x] = b1[threadIdx.x]; sb2[threadIdx.x] = b2[threadIdx.x]; }
  __syncthreads();
  int l = blockIdx.x * 256 + threadIdx.x;
  if (l >= L_N) return;
  float s = 0.0f;
  for (int i = 0; i < MPL; i++) s += traffic[ptl[(l * MPL + i) * 2]];
  float load = s / capacity[l];
  int pol = policy[l];
  float f[5] = {load, pol == 0 ? 1.f : 0.f, pol == 1 ? 1.f : 0.f,
                pol == 2 ? 1.f : 0.f, pol == 3 ? 1.f : 0.f};
  float h1[32];
#pragma unroll
  for (int j = 0; j < 32; j++) {
    float a = sb1[j];
    for (int i = 0; i < 5; i++) a += f[i] * sW1[i * 32 + j];
    h1[j] = fmaxf(a, 0.0f);
  }
#pragma unroll
  for (int j = 0; j < 32; j++) {
    float a = sb2[j];
    for (int i = 0; i < 32; i++) a += h1[i] * sW2[i * 32 + j];
    link_state[l * 32 + j] = fmaxf(a, 0.0f);
  }
}

__global__ void __launch_bounds__(256) queue_emb_kernel(
    const float* __restrict__ queue_size, const int* __restrict__ priority,
    const float* __restrict__ weight,
    const float* __restrict__ W1, const float* __restrict__ b1,
    const float* __restrict__ W2, const float* __restrict__ b2,
    float* __restrict__ queue_state) {
  __shared__ float sW1[5 * 32], sW2[32 * 32], sb1[32], sb2[32];
  for (int i = threadIdx.x; i < 5 * 32; i += 256) sW1[i] = W1[i];
  for (int i = threadIdx.x; i < 32 * 32; i += 256) sW2[i] = W2[i];
  if (threadIdx.x < 32) { sb1[threadIdx.x] = b1[threadIdx.x]; sb2[threadIdx.x] = b2[threadIdx.x]; }
  __syncthreads();
  int q = blockIdx.x * 256 + threadIdx.x;
  if (q >= Q_N) return;
  int pr = priority[q];
  float f[5] = {(queue_size[q] - 30259.10546875f) / 21410.095703125f,
                pr == 0 ? 1.f : 0.f, pr == 1 ? 1.f : 0.f, pr == 2 ? 1.f : 0.f,
                weight[q]};
  float h1[32];
#pragma unroll
  for (int j = 0; j < 32; j++) {
    float a = sb1[j];
    for (int i = 0; i < 5; i++) a += f[i] * sW1[i * 32 + j];
    h1[j] = fmaxf(a, 0.0f);
  }
#pragma unroll
  for (int j = 0; j < 32; j++) {
    float a = sb2[j];
    for (int i = 0; i < 32; i++) a += h1[i] * sW2[i * 32 + j];
    queue_state[q * 32 + j] = fmaxf(a, 0.0f);
  }
}

// ---------------------------------------------------------------------------
// Path GRU: one wave owns 16 paths for all T=8 steps. Weights live in VGPRs
// as B-fragments; h stays in registers in C-layout; LDS transposes h into an
// A-fragment each step. 18 v_wmma per step per wave. Next-step gather rows
// are prefetched (global_prefetch) to hide L2 latency behind WMMA+gates.
// ---------------------------------------------------------------------------
__global__ void __launch_bounds__(128) path_gru_kernel(
    const float* __restrict__ queue_state, const float* __restrict__ link_state,
    const int* __restrict__ qtp, const int* __restrict__ ltp,
    const int* __restrict__ length,
    const v16bf* __restrict__ wk, const v16bf* __restrict__ wrk,
    const float* __restrict__ bi, const float* __restrict__ bh,
    float* __restrict__ path_state, float* __restrict__ pss) {
  __shared__ float hbuf[4][16 * 33];
  int lane = threadIdx.x & 31;
  int wid  = threadIdx.x >> 5;
  int nlo = lane & 15;
  int khalf = lane >> 4;
  int mhi = khalf * 8;
  int pbase = (blockIdx.x * 4 + wid) * 16;

  v16bf kq[6], kl[6], rk[6];
  float bi6[6], bh6[6];
#pragma unroll
  for (int j = 0; j < 6; j++) {
    kq[j] = wk[j * 32 + lane];         // K-chunk 0 : queue half of x
    kl[j] = wk[(6 + j) * 32 + lane];   // K-chunk 1 : link half of x
    rk[j] = wrk[j * 32 + lane];
    bi6[j] = bi[16 * j + nlo];
    bh6[j] = bh[16 * j + nlo];
  }

  // this lane's timestep indices: 8 contiguous ints -> two b128 loads
  int mpath = pbase + nlo;
  int qis[8], lis[8];
  {
    const int4* qp = (const int4*)(qtp + mpath * T_N);
    const int4* lp = (const int4*)(ltp + mpath * T_N);
    int4 q01 = qp[0], q23 = qp[1], l01 = lp[0], l23 = lp[1];
    qis[0]=q01.x; qis[1]=q01.y; qis[2]=q01.z; qis[3]=q01.w;
    qis[4]=q23.x; qis[5]=q23.y; qis[6]=q23.z; qis[7]=q23.w;
    lis[0]=l01.x; lis[1]=l01.y; lis[2]=l01.z; lis[3]=l01.w;
    lis[4]=l23.x; lis[5]=l23.y; lis[6]=l23.z; lis[7]=l23.w;
  }
  __builtin_prefetch(queue_state + qis[0] * 32, 0, 3);
  __builtin_prefetch(link_state + lis[0] * 32, 0, 3);

  float h0[8], h1[8];
  int lenr[8];
#pragma unroll
  for (int r = 0; r < 8; r++) {
    int p = pbase + r + mhi;
    h0[r] = path_state[p * 32 + nlo];
    h1[r] = path_state[p * 32 + 16 + nlo];
    pss[p * 288 + nlo] = h0[r];          // path_state_sequence[:,0]
    pss[p * 288 + 16 + nlo] = h1[r];
    lenr[r] = length[p];
  }
  float* hb = hbuf[wid];

  for (int t = 0; t < T_N; t++) {
    // transpose h (C-layout regs) -> A-fragment via LDS
#pragma unroll
    for (int r = 0; r < 8; r++) {
      hb[(r + mhi) * 33 + nlo] = h0[r];
      hb[(r + mhi) * 33 + 16 + nlo] = h1[r];
    }
    __syncthreads();
    v16bf ah;
#pragma unroll
    for (int i = 0; i < 8; i++) {
      ah[i]     = (__bf16)hb[nlo * 33 + 8 * khalf + i];
      ah[8 + i] = (__bf16)hb[nlo * 33 + 16 + 8 * khalf + i];
    }
    __syncthreads();

    v16bf aq = gatherA(queue_state + qis[t] * 32, khalf);
    v16bf al = gatherA(link_state + lis[t] * 32, khalf);
    if (t + 1 < T_N) {   // overlap next step's L2 gather with this step's math
      __builtin_prefetch(queue_state + qis[t + 1] * 32, 0, 3);
      __builtin_prefetch(link_state + lis[t + 1] * 32, 0, 3);
    }

    v8f xg[6], hg[6];
#pragma unroll
    for (int j = 0; j < 6; j++) {
      v8f z = {0, 0, 0, 0, 0, 0, 0, 0};
      xg[j] = wmma_bf16(al, kl[j], wmma_bf16(aq, kq[j], z));
      hg[j] = wmma_bf16(ah, rk[j], z);
    }
#pragma unroll
    for (int r = 0; r < 8; r++) {
      int p = pbase + r + mhi;
      float o0, o1;
      gru_gates(xg, hg, bi6, bh6, r, h0[r], h1[r], o0, o1);
      if (t < lenr[r]) { h0[r] = o0; h1[r] = o1; }   // ragged mask
      pss[p * 288 + (t + 1) * 32 + nlo] = h0[r];
      pss[p * 288 + (t + 1) * 32 + 16 + nlo] = h1[r];
    }
  }
#pragma unroll
  for (int r = 0; r < 8; r++) {
    int p = pbase + r + mhi;
    path_state[p * 32 + nlo] = h0[r];
    path_state[p * 32 + 16 + nlo] = h1[r];
  }
}

// ---------------------------------------------------------------------------
// Queue GRU: 40-way gather-sum of path_state_sequence rows straight into an
// A-fragment, single GRU step with WMMA. No LDS needed.
// ---------------------------------------------------------------------------
__global__ void __launch_bounds__(128) queue_gru_kernel(
    const float* __restrict__ pss, const int* __restrict__ p2q,
    const v16bf* __restrict__ wk, const v16bf* __restrict__ wrk,
    const float* __restrict__ bi, const float* __restrict__ bh,
    float* __restrict__ queue_state) {
  int lane = threadIdx.x & 31;
  int wid = threadIdx.x >> 5;
  int qbase = (blockIdx.x * 4 + wid) * 16;
  if (qbase >= Q_N) return;
  int nlo = lane & 15, khalf = lane >> 4, mhi = khalf * 8;

  v16bf kk[6], rk[6];
  float bi6[6], bh6[6];
#pragma unroll
  for (int j = 0; j < 6; j++) {
    kk[j] = wk[j * 32 + lane];
    rk[j] = wrk[j * 32 + lane];
    bi6[j] = bi[16 * j + nlo];
    bh6[j] = bh[16 * j + nlo];
  }

  float acc[16];
#pragma unroll
  for (int i = 0; i < 16; i++) acc[i] = 0.0f;
  int q = qbase + nlo;
  for (int i = 0; i < MPQ; i++) {
    int pp  = p2q[(q * MPQ + i) * 2];
    int pos = p2q[(q * MPQ + i) * 2 + 1];
    const float* row = pss + pp * 288 + pos * 32;
    const float4* r1 = (const float4*)(row + 8 * khalf);
    const float4* r2 = (const float4*)(row + 16 + 8 * khalf);
    float4 a0 = r1[0], a1 = r1[1], b0 = r2[0], b1 = r2[1];
    acc[0] += a0.x; acc[1] += a0.y; acc[2] += a0.z; acc[3] += a0.w;
    acc[4] += a1.x; acc[5] += a1.y; acc[6] += a1.z; acc[7] += a1.w;
    acc[8] += b0.x; acc[9] += b0.y; acc[10] += b0.z; acc[11] += b0.w;
    acc[12] += b1.x; acc[13] += b1.y; acc[14] += b1.z; acc[15] += b1.w;
  }
  v16bf ax;
#pragma unroll
  for (int i = 0; i < 16; i++) ax[i] = (__bf16)acc[i];
  v16bf ah = gatherA(queue_state + q * 32, khalf);

  v8f xg[6], hg[6];
#pragma unroll
  for (int j = 0; j < 6; j++) {
    v8f z = {0, 0, 0, 0, 0, 0, 0, 0};
    xg[j] = wmma_bf16(ax, kk[j], z);
    hg[j] = wmma_bf16(ah, rk[j], z);
  }
#pragma unroll
  for (int r = 0; r < 8; r++) {
    int qq = qbase + r + mhi;
    float h0 = queue_state[qq * 32 + nlo];
    float h1 = queue_state[qq * 32 + 16 + nlo];
    float o0, o1;
    gru_gates(xg, hg, bi6, bh6, r, h0, h1, o0, o1);
    queue_state[qq * 32 + nlo] = o0;
    queue_state[qq * 32 + 16 + nlo] = o1;
  }
}

// ---------------------------------------------------------------------------
// Link GRU: 3 sequential GRU steps (one per queue slot). One wave per block,
// 16 links; h transposed through LDS per step.
// ---------------------------------------------------------------------------
__global__ void __launch_bounds__(32) link_gru_kernel(
    const float* __restrict__ queue_state, const int* __restrict__ qtl,
    const v16bf* __restrict__ wk, const v16bf* __restrict__ wrk,
    const float* __restrict__ bi, const float* __restrict__ bh,
    float* __restrict__ link_state) {
  __shared__ float hb[16 * 33];
  int lane = threadIdx.x & 31;
  int nlo = lane & 15, khalf = lane >> 4, mhi = khalf * 8;
  int lbase = blockIdx.x * 16;

  v16bf kk[6], rk[6];
  float bi6[6], bh6[6];
#pragma unroll
  for (int j = 0; j < 6; j++) {
    kk[j] = wk[j * 32 + lane];
    rk[j] = wrk[j * 32 + lane];
    bi6[j] = bi[16 * j + nlo];
    bh6[j] = bh[16 * j + nlo];
  }
  float h0[8], h1[8];
#pragma unroll
  for (int r = 0; r < 8; r++) {
    int l = lbase + r + mhi;
    h0[r] = link_state[l * 32 + nlo];
    h1[r] = link_state[l * 32 + 16 + nlo];
  }
  for (int j = 0; j < MQL; j++) {
#pragma unroll
    for (int r = 0; r < 8; r++) {
      hb[(r + mhi) * 33 + nlo] = h0[r];
      hb[(r + mhi) * 33 + 16 + nlo] = h1[r];
    }
    __syncthreads();
    v16bf ah;
#pragma unroll
    for (int i = 0; i < 8; i++) {
      ah[i]     = (__bf16)hb[nlo * 33 + 8 * khalf + i];
      ah[8 + i] = (__bf16)hb[nlo * 33 + 16 + 8 * khalf + i];
    }
    __syncthreads();
    int q = qtl[(lbase + nlo) * MQL + j];
    v16bf ax = gatherA(queue_state + q * 32, khalf);
    v8f xg[6], hg[6];
#pragma unroll
    for (int jj = 0; jj < 6; jj++) {
      v8f z = {0, 0, 0, 0, 0, 0, 0, 0};
      xg[jj] = wmma_bf16(ax, kk[jj], z);
      hg[jj] = wmma_bf16(ah, rk[jj], z);
    }
#pragma unroll
    for (int r = 0; r < 8; r++) {
      float o0, o1;
      gru_gates(xg, hg, bi6, bh6, r, h0[r], h1[r], o0, o1);
      h0[r] = o0; h1[r] = o1;
    }
  }
#pragma unroll
  for (int r = 0; r < 8; r++) {
    int l = lbase + r + mhi;
    link_state[l * 32 + nlo] = h0[r];
    link_state[l * 32 + 16 + nlo] = h1[r];
  }
}

// ---------------------------------------------------------------------------
// Readout MLP + delay reduction (per-path thread, weights in LDS)
// ---------------------------------------------------------------------------
__global__ void __launch_bounds__(256) readout_kernel(
    const float* __restrict__ pss, const float* __restrict__ capacity,
    const int* __restrict__ ltp, const int* __restrict__ length,
    const float* __restrict__ traffic, const float* __restrict__ packets,
    const float* __restrict__ W1, const float* __restrict__ b1,
    const float* __restrict__ W2, const float* __restrict__ b2,
    const float* __restrict__ W3, const float* __restrict__ b3,
    float* __restrict__ out) {
  __shared__ float sW1[32 * 16], sW2[16 * 16], sb1[16], sb2[16], sW3[16], sb3;
  for (int i = threadIdx.x; i < 32 * 16; i += 256) sW1[i] = W1[i];
  for (int i = threadIdx.x; i < 16 * 16; i += 256) sW2[i] = W2[i];
  if (threadIdx.x < 16) {
    sb1[threadIdx.x] = b1[threadIdx.x];
    sb2[threadIdx.x] = b2[threadIdx.x];
    sW3[threadIdx.x] = W3[threadIdx.x];
  }
  if (threadIdx.x == 0) sb3 = b3[0];
  __syncthreads();
  int p = blockIdx.x * 256 + threadIdx.x;
  if (p >= P_N) return;
  int len = length[p];
  float qd = 0.0f, ic = 0.0f;
  for (int t = 0; t < T_N; t++) {
    if (t >= len) continue;
    const float* row = pss + p * 288 + (t + 1) * 32;
    float o1[16];
#pragma unroll
    for (int j = 0; j < 16; j++) {
      float s = sb1[j];
      for (int i = 0; i < 32; i++) s += row[i] * sW1[i * 16 + j];
      o1[j] = fmaxf(s, 0.0f);
    }
    float o2[16];
#pragma unroll
    for (int j = 0; j < 16; j++) {
      float s = sb2[j];
      for (int i = 0; i < 16; i++) s += o1[i] * sW2[i * 16 + j];
      o2[j] = fmaxf(s, 0.0f);
    }
    float occ = sb3;
#pragma unroll
    for (int i = 0; i < 16; i++) occ += o2[i] * sW3[i];
    float rc = __builtin_amdgcn_rcpf(capacity[ltp[p * T_N + t]]);
    qd += occ * rc;
    ic += rc;
  }
  out[p] = qd + (traffic[p] / packets[p]) * ic;
}

// ---------------------------------------------------------------------------
enum {
  IN_TRAFFIC = 0, IN_PACKETS, IN_EQL, IN_APL, IN_EMF, IN_PLO, IN_TOFF, IN_TON,
  IN_ARA, IN_SIGMA, IN_CAP, IN_QSIZE, IN_WEIGHT, IN_MODEL, IN_POLICY,
  IN_PRIORITY, IN_LENGTH, IN_QTP, IN_LTP, IN_PTL, IN_QTL, IN_P2Q,
  IN_PE_W1, IN_PE_B1, IN_PE_W2, IN_PE_B2,
  IN_LE_W1, IN_LE_B1, IN_LE_W2, IN_LE_B2,
  IN_QE_W1, IN_QE_B1, IN_QE_W2, IN_QE_B2,
  IN_PG_K, IN_PG_RK, IN_PG_BI, IN_PG_BH,
  IN_QG_K, IN_QG_RK, IN_QG_BI, IN_QG_BH,
  IN_LG_K, IN_LG_RK, IN_LG_BI, IN_LG_BH,
  IN_RO_W1, IN_RO_B1, IN_RO_W2, IN_RO_B2, IN_RO_W3, IN_RO_B3
};

extern "C" void kernel_launch(void* const* d_in, const int* in_sizes, int n_in,
                              void* d_out, int out_size, void* d_ws, size_t ws_size,
                              hipStream_t stream) {
  (void)in_sizes; (void)n_in; (void)out_size; (void)ws_size;
  char* ws = (char*)d_ws;
  size_t off = 0;
  auto alloc = [&](size_t bytes) -> char* {
    char* p = ws + off;
    off += (bytes + 255) & ~(size_t)255;
    return p;
  };
  float* path_state  = (float*)alloc((size_t)P_N * 32 * 4);
  float* pss         = (float*)alloc((size_t)P_N * 288 * 4);  // [P, T+1, D]
  float* link_state  = (float*)alloc((size_t)L_N * 32 * 4);
  float* queue_state = (float*)alloc((size_t)Q_N * 32 * 4);
  unsigned int* wpk  = (unsigned int*)alloc(2 * 6 * 32 * 8 * 4);
  unsigned int* wprk = (unsigned int*)alloc(6 * 32 * 8 * 4);
  unsigned int* wqk  = (unsigned int*)alloc(6 * 32 * 8 * 4);
  unsigned int* wqrk = (unsigned int*)alloc(6 * 32 * 8 * 4);
  unsigned int* wlk  = (unsigned int*)alloc(6 * 32 * 8 * 4);
  unsigned int* wlrk = (unsigned int*)alloc(6 * 32 * 8 * 4);

  // pack GRU weights into bf16 B-fragments
  pack_b_kernel<<<12, 256, 0, stream>>>((const float*)d_in[IN_PG_K],  wpk, 2);
  pack_b_kernel<<<6, 256, 0, stream>>>((const float*)d_in[IN_PG_RK], wprk, 1);
  pack_b_kernel<<<6, 256, 0, stream>>>((const float*)d_in[IN_QG_K],  wqk, 1);
  pack_b_kernel<<<6, 256, 0, stream>>>((const float*)d_in[IN_QG_RK], wqrk, 1);
  pack_b_kernel<<<6, 256, 0, stream>>>((const float*)d_in[IN_LG_K],  wlk, 1);
  pack_b_kernel<<<6, 256, 0, stream>>>((const float*)d_in[IN_LG_RK], wlrk, 1);

  // embeddings
  path_emb_kernel<<<(P_N + 255) / 256, 256, 0, stream>>>(
      (const float*)d_in[IN_TRAFFIC], (const float*)d_in[IN_PACKETS],
      (const float*)d_in[IN_EQL], (const float*)d_in[IN_APL],
      (const float*)d_in[IN_EMF], (const float*)d_in[IN_PLO],
      (const float*)d_in[IN_TOFF], (const float*)d_in[IN_TON],
      (const float*)d_in[IN_ARA], (const float*)d_in[IN_SIGMA],
      (const int*)d_in[IN_MODEL],
      (const float*)d_in[IN_PE_W1], (const float*)d_in[IN_PE_B1],
      (const float*)d_in[IN_PE_W2], (const float*)d_in[IN_PE_B2], path_state);
  link_emb_kernel<<<(L_N + 255) / 256, 256, 0, stream>>>(
      (const float*)d_in[IN_TRAFFIC], (const float*)d_in[IN_CAP],
      (const int*)d_in[IN_POLICY], (const int*)d_in[IN_PTL],
      (const float*)d_in[IN_LE_W1], (const float*)d_in[IN_LE_B1],
      (const float*)d_in[IN_LE_W2], (const float*)d_in[IN_LE_B2], link_state);
  queue_emb_kernel<<<(Q_N + 255) / 256, 256, 0, stream>>>(
      (const float*)d_in[IN_QSIZE], (const int*)d_in[IN_PRIORITY],
      (const float*)d_in[IN_WEIGHT],
      (const float*)d_in[IN_QE_W1], (const float*)d_in[IN_QE_B1],
      (const float*)d_in[IN_QE_W2], (const float*)d_in[IN_QE_B2], queue_state);

  // message-passing iterations
  for (int it = 0; it < ITERS; ++it) {
    path_gru_kernel<<<P_N / 64, 128, 0, stream>>>(
        queue_state, link_state, (const int*)d_in[IN_QTP],
        (const int*)d_in[IN_LTP], (const int*)d_in[IN_LENGTH],
        (const v16bf*)wpk, (const v16bf*)wprk,
        (const float*)d_in[IN_PG_BI], (const float*)d_in[IN_PG_BH],
        path_state, pss);
    queue_gru_kernel<<<(Q_N / 16 + 3) / 4, 128, 0, stream>>>(
        pss, (const int*)d_in[IN_P2Q], (const v16bf*)wqk, (const v16bf*)wqrk,
        (const float*)d_in[IN_QG_BI], (const float*)d_in[IN_QG_BH], queue_state);
    link_gru_kernel<<<L_N / 16, 32, 0, stream>>>(
        queue_state, (const int*)d_in[IN_QTL], (const v16bf*)wlk,
        (const v16bf*)wlrk, (const float*)d_in[IN_LG_BI],
        (const float*)d_in[IN_LG_BH], link_state);
  }

  // readout + delay reduction
  readout_kernel<<<(P_N + 255) / 256, 256, 0, stream>>>(
      pss, (const float*)d_in[IN_CAP], (const int*)d_in[IN_LTP],
      (const int*)d_in[IN_LENGTH], (const float*)d_in[IN_TRAFFIC],
      (const float*)d_in[IN_PACKETS],
      (const float*)d_in[IN_RO_W1], (const float*)d_in[IN_RO_B1],
      (const float*)d_in[IN_RO_W2], (const float*)d_in[IN_RO_B2],
      (const float*)d_in[IN_RO_W3], (const float*)d_in[IN_RO_B3],
      (float*)d_out);
}
